// opening_critic_23570780520494
// MI455X (gfx1250) — matrix operand order repaired and checked by reading, not compiled, not run on hardware
//
#include <hip/hip_runtime.h>

typedef __attribute__((ext_vector_type(16))) _Float16 v16h;
typedef __attribute__((ext_vector_type(8)))  float    v8f;

#define HPF 512  // halves per B-fragment: 32 lanes * 16 halves

// fragment offsets in d_ws (halves)
enum : int {
  FR_RW1 = 0 * HPF,    // 135->64 : 5 ksteps * 4 ntiles = 20 frags
  FR_KW1 = 20 * HPF,   // 128->64 : 16
  FR_RW2 = 36 * HPF,   // 64->32  : 4
  FR_KW2 = 40 * HPF,   // 64->32  : 4
  FR_RW3 = 44 * HPF,   // 32->16  : 1
  FR_AW1 = 45 * HPF,   // 32->32  : 2
  FR_AW2 = 47 * HPF,   // 32->16  : 1
  FR_GW1 = 48 * HPF,   // 32->16  : 1
  FR_QW1 = 49 * HPF,   // 6->16   : 1
  FR_QW2 = 50 * HPF,   // 16->32  : 2
  FR_VW1 = 52 * HPF,   // 1->16   : 1
  FR_VW2 = 54 * HPF,   // 16->32  : 2
  FR_GW2 = 55 * HPF    // 16->1   : 1
};

// Per-wave LDS layout (halves): X[32][160] HA[32][64] HB,HC,HD[32][32] RES[32][16]
constexpr int W_HALVES = 32 * 160 + 32 * 64 + 3 * 32 * 32 + 32 * 16; // 10752 halves

// ---------------------------------------------------------------------------
// Pack fp32 weight (K x Nf, row-major) into WMMA 16x16x32 f16 B fragments.
// B layout (wave32): lane L -> col n = L&15 within n-tile; halves i=0..15 hold
// k = kstep*32 + (L>>4)*16 + i. Zero-pad outside [K, Nf].
// ---------------------------------------------------------------------------
__global__ void pack_weights(const float* __restrict__ W, int K, int Nf,
                             int ksteps, int ntiles, _Float16* __restrict__ dst)
{
  int t = blockIdx.x * blockDim.x + threadIdx.x;
  int nfrag = ksteps * ntiles;
  if (t >= nfrag * 32) return;
  int frag = t >> 5, lane = t & 31;
  int nt = frag % ntiles, s = frag / ntiles;
  int n  = nt * 16 + (lane & 15);
  int kb = s * 32 + (lane >> 4) * 16;
  _Float16* d = dst + (size_t)(frag * 32 + lane) * 16;
#pragma unroll
  for (int i = 0; i < 16; ++i) {
    int k = kb + i;
    float v = (k < K && n < Nf) ? W[(size_t)k * Nf + n] : 0.f;
    d[i] = (_Float16)v;
  }
}

// ---------------------------------------------------------------------------
// One 16x16 output tile: accumulate over ksteps K-slices of 32.
// A layout (16-bit, 16x32): lane L holds row m = L&15; halves i=0..15 -> K =
// base + (L>>4)*16 + i.
// ---------------------------------------------------------------------------
__device__ __forceinline__ v8f wmma_kloop(const _Float16* xbase, int strideX,
                                          const _Float16* __restrict__ wfr,
                                          int ksteps, int ntiles, int nt, int lane)
{
  const int mlo = lane & 15, hi = lane >> 4;
  v8f acc = {};
  for (int s = 0; s < ksteps; ++s) {
    const _Float16* ap = xbase + mlo * strideX + s * 32 + hi * 16;
    v16h a;
#pragma unroll
    for (int i = 0; i < 8; ++i) { a[i] = ap[i]; a[8 + i] = ap[8 + i]; }
    const v16h b = *(const v16h*)(wfr + (size_t)((s * ntiles + nt) * 32 + lane) * 16);
    acc = __builtin_amdgcn_wmma_f32_16x16x32_f16(false, a, false, b,
                                                 (short)0, acc, false, false);
  }
  return acc;
}

// Y[32][Fout] = X[32][K] @ W (+bias), stored as f16 into LDS.
// D layout: lane L holds col n = L&15; VGPR j -> row m = mt*16 + (L>>4)*8 + j.
__device__ __forceinline__ void wave_gemm(const _Float16* X, int sX,
                                          const _Float16* __restrict__ wfr,
                                          int ksteps, int ntiles,
                                          _Float16* Y, int sY,
                                          const float* __restrict__ bias, int lane)
{
  const int mlo = lane & 15, hi = lane >> 4;
  for (int mt = 0; mt < 2; ++mt)
    for (int nt = 0; nt < ntiles; ++nt) {
      v8f acc = wmma_kloop(X + mt * 16 * sX, sX, wfr, ksteps, ntiles, nt, lane);
      const float bv = bias ? bias[nt * 16 + mlo] : 0.f;
#pragma unroll
      for (int j = 0; j < 8; ++j) {
        const int m = mt * 16 + hi * 8 + j;
        Y[m * sY + nt * 16 + mlo] = (_Float16)(acc[j] + bv);
      }
    }
}

// LayerNorm + affine + ReLU, one lane per row (32 rows per wave).
__device__ __forceinline__ void wave_ln_relu(_Float16* Y, int sY, int F,
                                             const float* __restrict__ g,
                                             const float* __restrict__ b, int lane)
{
  _Float16* row = Y + lane * sY;
  float s = 0.f, s2 = 0.f;
  for (int f = 0; f < F; ++f) { float x = (float)row[f]; s += x; s2 += x * x; }
  const float invF = 1.f / (float)F;
  const float m = s * invF;
  const float var = fmaxf(s2 * invF - m * m, 0.f);
  const float rs = rsqrtf(var + 1e-5f);
  for (int f = 0; f < F; ++f) {
    float x = ((float)row[f] - m) * rs * g[f] + b[f];
    row[f] = (_Float16)fmaxf(x, 0.f);
  }
}

struct Params {
  const float *rep, *pose, *open_;
  const float *kg, *kb, *kb2;
  const float *qg, *qb, *qb2;
  const float *vg, *vb, *vb2;
  const float *ag, *ab, *ab2;
  const float *rg1, *rb1, *rg2, *rb2, *rb3;
  const float *gg, *gb, *gb2;
  const _Float16* wf;
  float* out;
  int n;
};

__global__ __launch_bounds__(64) void opening_critic_main(Params p)
{
  __shared__ _Float16 smem[2 * W_HALVES];  // 43008 bytes, 2 waves per block
  const int tid  = threadIdx.x;
  const int lane = tid & 31;
  const int wv   = tid >> 5;
  const int blkRow0 = blockIdx.x * 64;

  // ---- Phase 0: cooperative load of 64 rows into X tiles (f16) -------------
  // cols 0..127 = rep, 128..133 = pose, 134 = opening, 135..159 = zero pad
  for (int i = tid; i < 64 * 128; i += 64) {
    int r = i >> 7, c = i & 127, gr = blkRow0 + r;
    float v = (gr < p.n) ? p.rep[(size_t)gr * 128 + c] : 0.f;
    smem[(r >> 5) * W_HALVES + (r & 31) * 160 + c] = (_Float16)v;
  }
  for (int i = tid; i < 64 * 6; i += 64) {
    int r = i / 6, c = i % 6, gr = blkRow0 + r;
    float v = (gr < p.n) ? p.pose[(size_t)gr * 6 + c] : 0.f;
    smem[(r >> 5) * W_HALVES + (r & 31) * 160 + 128 + c] = (_Float16)v;
  }
  for (int i = tid; i < 64 * 26; i += 64) {
    int r = i / 26, c = i % 26, gr = blkRow0 + r;
    float v = (c == 0 && gr < p.n) ? p.open_[gr] : 0.f;
    smem[(r >> 5) * W_HALVES + (r & 31) * 160 + 134 + c] = (_Float16)v;
  }
  __syncthreads();

  _Float16* X   = smem + wv * W_HALVES;
  _Float16* HA  = X  + 32 * 160;
  _Float16* HB  = HA + 32 * 64;
  _Float16* HC  = HB + 32 * 32;
  _Float16* HD  = HC + 32 * 32;
  _Float16* RES = HD + 32 * 32;

  // ---- residual path: 135 ->64 ->32 ->16 -----------------------------------
  wave_gemm(X, 160, p.wf + FR_RW1, 5, 4, HA, 64, nullptr, lane);
  wave_ln_relu(HA, 64, 64, p.rg1, p.rb1, lane);
  wave_gemm(HA, 64, p.wf + FR_RW2, 2, 2, HB, 32, nullptr, lane);
  wave_ln_relu(HB, 32, 32, p.rg2, p.rb2, lane);
  wave_gemm(HB, 32, p.wf + FR_RW3, 1, 1, RES, 16, p.rb3, lane);

  // ---- key branch: 128 ->64 ->32 ------------------------------------------
  wave_gemm(X, 160, p.wf + FR_KW1, 4, 4, HA, 64, nullptr, lane);
  wave_ln_relu(HA, 64, 64, p.kg, p.kb, lane);
  wave_gemm(HA, 64, p.wf + FR_KW2, 2, 2, HB, 32, p.kb2, lane);     // key -> HB

  // ---- query branch: 6 ->16 ->32 (K zero-padded to 32 in packed weights) ---
  wave_gemm(X + 128, 160, p.wf + FR_QW1, 1, 1, HA, 64, nullptr, lane);
  wave_ln_relu(HA, 64, 16, p.qg, p.qb, lane);
  wave_gemm(HA, 64, p.wf + FR_QW2, 1, 2, HC, 32, p.qb2, lane);     // query -> HC

  // ---- value branch: 1 ->16 ->32 ------------------------------------------
  wave_gemm(X + 134, 160, p.wf + FR_VW1, 1, 1, HA, 64, nullptr, lane);
  wave_ln_relu(HA, 64, 16, p.vg, p.vb, lane);
  wave_gemm(HA, 64, p.wf + FR_VW2, 1, 2, HD, 32, p.vb2, lane);     // value -> HD

  // ---- att_in = softmax(query*key) * value (one lane per row) -> HB --------
  {
    _Float16* k = HB + lane * 32;
    _Float16* q = HC + lane * 32;
    _Float16* v = HD + lane * 32;
    float t[32];
    float mx = -1e30f;
#pragma unroll
    for (int f = 0; f < 32; ++f) { t[f] = (float)q[f] * (float)k[f]; mx = fmaxf(mx, t[f]); }
    float sum = 0.f;
#pragma unroll
    for (int f = 0; f < 32; ++f) { t[f] = __expf(t[f] - mx); sum += t[f]; }
    const float inv = 1.f / sum;
#pragma unroll
    for (int f = 0; f < 32; ++f) k[f] = (_Float16)(t[f] * inv * (float)v[f]);
  }

  // ---- attention head: 32 ->32 ->16 ----------------------------------------
  wave_gemm(HB, 32, p.wf + FR_AW1, 1, 2, HC, 32, nullptr, lane);
  wave_ln_relu(HC, 32, 32, p.ag, p.ab, lane);
  wave_gemm(HC, 32, p.wf + FR_AW2, 1, 1, HB, 32, p.ab2, lane);     // att -> HB[:,0:16]

  // features = concat(att, residuals) -> HB[:,0:32]
#pragma unroll
  for (int f = 0; f < 16; ++f)
    HB[lane * 32 + 16 + f] = RES[lane * 16 + f];

  // ---- get_value head: 32 ->16 ->1 -----------------------------------------
  wave_gemm(HB, 32, p.wf + FR_GW1, 1, 1, HC, 32, nullptr, lane);
  wave_ln_relu(HC, 32, 16, p.gg, p.gb, lane);

  {
    const int mlo = lane & 15, hi = lane >> 4;
    const float gb2v = p.gb2[0];
    for (int mt = 0; mt < 2; ++mt) {
      v8f acc = wmma_kloop(HC + mt * 16 * 32, 32, p.wf + FR_GW2, 1, 1, 0, lane);
      if (mlo == 0) {  // only output column n==0 is valid
#pragma unroll
        for (int j = 0; j < 8; ++j) {
          int m  = mt * 16 + hi * 8 + j;
          int gr = blkRow0 + wv * 32 + m;
          if (gr < p.n) p.out[gr] = acc[j] + gb2v;
        }
      }
    }
  }
}

extern "C" void kernel_launch(void* const* d_in, const int* in_sizes, int n_in,
                              void* d_out, int out_size, void* d_ws, size_t ws_size,
                              hipStream_t stream)
{
  (void)n_in; (void)out_size; (void)ws_size;
  _Float16* wf = (_Float16*)d_ws;

  auto pack = [&](int idx, int K, int Nf, int ks, int nt, int off) {
    int threads = ks * nt * 32;
    pack_weights<<<(threads + 63) / 64, 64, 0, stream>>>(
        (const float*)d_in[idx], K, Nf, ks, nt, wf + off);
  };
  pack(23, 135, 64, 5, 4, FR_RW1);
  pack( 3, 128, 64, 4, 4, FR_KW1);
  pack(26,  64, 32, 2, 2, FR_RW2);
  pack( 6,  64, 32, 2, 2, FR_KW2);
  pack(29,  32, 16, 1, 1, FR_RW3);
  pack(18,  32, 32, 1, 2, FR_AW1);
  pack(21,  32, 16, 1, 1, FR_AW2);
  pack(31,  32, 16, 1, 1, FR_GW1);
  pack( 8,   6, 16, 1, 1, FR_QW1);
  pack(11,  16, 32, 1, 2, FR_QW2);
  pack(13,   1, 16, 1, 1, FR_VW1);
  pack(16,  16, 32, 1, 2, FR_VW2);
  pack(34,  16,  1, 1, 1, FR_GW2);

  Params p;
  p.rep  = (const float*)d_in[0];
  p.pose = (const float*)d_in[1];
  p.open_= (const float*)d_in[2];
  p.kg  = (const float*)d_in[4];  p.kb  = (const float*)d_in[5];  p.kb2 = (const float*)d_in[7];
  p.qg  = (const float*)d_in[9];  p.qb  = (const float*)d_in[10]; p.qb2 = (const float*)d_in[12];
  p.vg  = (const float*)d_in[14]; p.vb  = (const float*)d_in[15]; p.vb2 = (const float*)d_in[17];
  p.ag  = (const float*)d_in[19]; p.ab  = (const float*)d_in[20]; p.ab2 = (const float*)d_in[22];
  p.rg1 = (const float*)d_in[24]; p.rb1 = (const float*)d_in[25];
  p.rg2 = (const float*)d_in[27]; p.rb2 = (const float*)d_in[28];
  p.rb3 = (const float*)d_in[30];
  p.gg  = (const float*)d_in[32]; p.gb  = (const float*)d_in[33]; p.gb2 = (const float*)d_in[35];
  p.wf  = wf;
  p.out = (float*)d_out;
  p.n   = in_sizes[0] / 128;

  opening_critic_main<<<(p.n + 63) / 64, 64, 0, stream>>>(p);
}
